// GlobalSAModule_68410239091222
// MI455X (gfx1250) — matrix operand order repaired and checked by reading, not compiled, not run on hardware
//
#include <hip/hip_runtime.h>
#include <hip/hip_bf16.h>

typedef __attribute__((ext_vector_type(16))) _Float16 v16h;
typedef __attribute__((ext_vector_type(8)))  _Float16 v8h;
typedef __attribute__((ext_vector_type(8)))  float    v8f;

#define TILE_M   128
#define K1P      288   // 259 padded to multiple of 32
#define N1       256
#define N2       512
#define N3       1024
#define NPTS     4096
#define NROWS    65536
#define NB       4     // N-tiles blocked per wave (A-fragment reuse)

// ---------------------------------------------------------------------------
// Fragment loaders (wave32, 16x16x32 f16 WMMA layouts per CDNA5 ISA 7.12.2)
// A (16x32, MxK): lane L<16 -> row M=L, K {0..7,16..23}; lane L+16 -> K {8..15,24..31}
// B (32x16, KxN): lane L<16 -> col N=L, K {0..7,16..23}; lane L+16 -> col N=L, K {8..15,24..31}
// C/D (16x16 f32): lane L -> col N=L&15 ; vgpr j -> row M = (L>=16?8:0)+j
// ---------------------------------------------------------------------------
__device__ __forceinline__ v16h load_fragA_lds(const _Float16* A, int ldk,
                                               int m0, int kb, int lane) {
  const int h = lane >> 4;
  const int m = lane & 15;
  const _Float16* p = A + (m0 + m) * ldk + kb + h * 8;
  v8h lo = *(const v8h*)(p);        // K kb+h*8   .. +8   -> vgprs 0..3
  v8h hi = *(const v8h*)(p + 16);   // K kb+16+h*8.. +8   -> vgprs 4..7
  v16h r;
#pragma unroll
  for (int i = 0; i < 8; ++i) { r[i] = lo[i]; r[i + 8] = hi[i]; }
  return r;
}

__device__ __forceinline__ v16h load_fragB_glb(const _Float16* __restrict__ Wt,
                                               int Kp, int n0, int kb, int lane) {
  const int h = lane >> 4;
  const int n = lane & 15;
  const _Float16* p = Wt + (size_t)(n0 + n) * Kp + kb + h * 8;
  v8h lo = *(const v8h*)(p);        // K kb+h*8     .. +8
  v8h hi = *(const v8h*)(p + 16);   // K kb+16+h*8  .. +8
  v16h r;
#pragma unroll
  for (int i = 0; i < 8; ++i) { r[i] = lo[i]; r[i + 8] = hi[i]; }
  return r;
}

__device__ __forceinline__ void store_tile_relu_lds(v8f acc, _Float16* H, int ldn,
                                                    int m0, int n0, int lane) {
  const int h = lane >> 4;
  const int n = lane & 15;
#pragma unroll
  for (int j = 0; j < 8; ++j) {
    float v = acc[j];
    v = v > 0.f ? v : 0.f;
    H[(m0 + h * 8 + j) * ldn + (n0 + n)] = (_Float16)v;
  }
}

// ---------------------------------------------------------------------------
// Fused 3-layer MLP + per-graph max pool. One block = 128 rows (one graph).
// LDS: Xtile 128x288 f16 | h1 128x256 f16 | h2 128x512 f16 | pool 1024 u32
//    = 73728 + 65536 + 131072 + 4096 = 274,432 B  (fits 320KB WGP LDS)
// The sH2 region doubles as the async-DMA staging buffer for the f32 X tile
// (128x256 f32 = 131072 B, exactly the sH2 footprint) before GEMM2 needs it.
// ---------------------------------------------------------------------------
__global__ __launch_bounds__(256) void fused_mlp_pool(
    const float* __restrict__ x, const float* __restrict__ pos,
    const _Float16* __restrict__ W1t, const float* __restrict__ b1,
    const _Float16* __restrict__ W2t, const float* __restrict__ b2,
    const _Float16* __restrict__ W3t, const float* __restrict__ b3,
    unsigned* __restrict__ pooledBits) {
  extern __shared__ char smem[];
  _Float16* sX  = (_Float16*)smem;                 // [128][288]
  _Float16* sH1 = sX  + TILE_M * K1P;              // [128][256]
  _Float16* sH2 = sH1 + TILE_M * N1;               // [128][512]
  unsigned* sPool = (unsigned*)(sH2 + TILE_M * N2); // [1024]

  const int tid  = threadIdx.x;
  const int wave = tid >> 5;
  const int lane = tid & 31;
  const int ln   = lane & 15;
  const int m0   = wave * 16;                      // each wave owns 16 rows
  const int rowBase = blockIdx.x * TILE_M;
  const int graph   = rowBase / NPTS;

  for (int i = tid; i < N3; i += 256) sPool[i] = 0u;

  // ---- Async DMA: contiguous 128KB f32 x-tile -> LDS (sH2 staging) --------
  // GLOBAL_LOAD_ASYNC_TO_LDS_B128: each lane copies 16 B from global to LDS, tracked by ASYNCcnt.
  {
    const char* gsrc = (const char*)(x + (size_t)rowBase * 256);
    const unsigned ldsBase = (unsigned)(uintptr_t)sH2;  // addr[31:0] == LDS offset
#pragma unroll 4
    for (int it = 0; it < 32; ++it) {                   // 8192 x 16B chunks total
      const int ci = tid + it * 256;
      const unsigned loff = ldsBase + ci * 16;
      const char* g = gsrc + (size_t)ci * 16;
      asm volatile("global_load_async_to_lds_b128 %0, %1, off"
                   :: "v"(loff), "v"(g) : "memory");
    }
    asm volatile("s_wait_asynccnt 0x0" ::: "memory");
  }
  __syncthreads();

  // ---- Convert staged f32 -> f16 into sX, append pos cols + zero pad ------
  {
    const float* sXf = (const float*)sH2;
    for (int i = tid; i < TILE_M * 64; i += 256) {      // x cols, float4 chunks
      const int r = i >> 6, c = (i & 63) * 4;
      const float4 v = *(const float4*)(sXf + r * 256 + c);
      _Float16* d = sX + r * K1P + c;
      d[0] = (_Float16)v.x; d[1] = (_Float16)v.y;
      d[2] = (_Float16)v.z; d[3] = (_Float16)v.w;
    }
    for (int i = tid; i < TILE_M * 32; i += 256) {      // pos cols + padding
      const int r = i >> 5, c = 256 + (i & 31);
      float v = (c < 259) ? pos[(size_t)(rowBase + r) * 3 + (c - 256)] : 0.f;
      sX[r * K1P + c] = (_Float16)v;
    }
  }
  __syncthreads();

  // ---- GEMM1: (128x288) @ (288x256) + b1, ReLU -> sH1 ---------------------
#pragma unroll 1
  for (int g = 0; g < N1 / (16 * NB); ++g) {
    const int n0 = g * 16 * NB;
    v8f acc[NB];
#pragma unroll
    for (int q = 0; q < NB; ++q) {
      const float bias = b1[n0 + q * 16 + ln];
#pragma unroll
      for (int j = 0; j < 8; ++j) acc[q][j] = bias;
    }
#pragma unroll
    for (int kk = 0; kk < K1P; kk += 32) {
      const v16h a = load_fragA_lds(sX, K1P, m0, kk, lane);
#pragma unroll
      for (int q = 0; q < NB; ++q) {
        const v16h b = load_fragB_glb(W1t, K1P, n0 + q * 16, kk, lane);
        acc[q] = __builtin_amdgcn_wmma_f32_16x16x32_f16(false, a, false, b,
                                                        (short)0, acc[q], false, false);
      }
    }
#pragma unroll
    for (int q = 0; q < NB; ++q)
      store_tile_relu_lds(acc[q], sH1, N1, m0, n0 + q * 16, lane);
  }
  __syncthreads();

  // ---- GEMM2: (128x256) @ (256x512) + b2, ReLU -> sH2 ---------------------
#pragma unroll 1
  for (int g = 0; g < N2 / (16 * NB); ++g) {
    const int n0 = g * 16 * NB;
    v8f acc[NB];
#pragma unroll
    for (int q = 0; q < NB; ++q) {
      const float bias = b2[n0 + q * 16 + ln];
#pragma unroll
      for (int j = 0; j < 8; ++j) acc[q][j] = bias;
    }
#pragma unroll
    for (int kk = 0; kk < N1; kk += 32) {
      const v16h a = load_fragA_lds(sH1, N1, m0, kk, lane);
#pragma unroll
      for (int q = 0; q < NB; ++q) {
        const v16h b = load_fragB_glb(W2t, N1, n0 + q * 16, kk, lane);
        acc[q] = __builtin_amdgcn_wmma_f32_16x16x32_f16(false, a, false, b,
                                                        (short)0, acc[q], false, false);
      }
    }
#pragma unroll
    for (int q = 0; q < NB; ++q)
      store_tile_relu_lds(acc[q], sH2, N2, m0, n0 + q * 16, lane);
  }
  __syncthreads();

  // ---- GEMM3: (128x512) @ (512x1024) + b3, ReLU fused into col-max --------
#pragma unroll 1
  for (int g = 0; g < N3 / (16 * NB); ++g) {
    const int n0 = g * 16 * NB;
    v8f acc[NB];
#pragma unroll
    for (int q = 0; q < NB; ++q) {
      const float bias = b3[n0 + q * 16 + ln];
#pragma unroll
      for (int j = 0; j < 8; ++j) acc[q][j] = bias;
    }
#pragma unroll 4
    for (int kk = 0; kk < N2; kk += 32) {
      const v16h a = load_fragA_lds(sH2, N2, m0, kk, lane);
#pragma unroll
      for (int q = 0; q < NB; ++q) {
        const v16h b = load_fragB_glb(W3t, N2, n0 + q * 16, kk, lane);
        acc[q] = __builtin_amdgcn_wmma_f32_16x16x32_f16(false, a, false, b,
                                                        (short)0, acc[q], false, false);
      }
    }
    // ReLU + max over this tile's 16 rows: start at 0 (== relu floor),
    // reduce 8 rows in-lane, fold lanes l <-> l+16 (same output column).
#pragma unroll
    for (int q = 0; q < NB; ++q) {
      float cm = 0.f;
#pragma unroll
      for (int j = 0; j < 8; ++j) cm = fmaxf(cm, acc[q][j]);
      cm = fmaxf(cm, __shfl_xor(cm, 16, 32));
      if (lane < 16) atomicMax(&sPool[n0 + q * 16 + lane], __float_as_uint(cm));
    }
  }
  __syncthreads();

  // one global atomic per (graph, col) per block — relu>=0 so uint-order == float-order
  for (int i = tid; i < N3; i += 256)
    atomicMax(&pooledBits[(size_t)graph * N3 + i], sPool[i]);
}

// ---------------------------------------------------------------------------
// Prep: transpose f32 weight [K][Nfull] -> f16 [Nfull][Kp] (zero-padded K)
// ---------------------------------------------------------------------------
__global__ void transpose_pad_f16(const float* __restrict__ W,
                                  _Float16* __restrict__ Wt,
                                  int K, int Nfull, int Kp) {
  const int idx = blockIdx.x * 256 + threadIdx.x;
  if (idx >= Nfull * Kp) return;
  const int n = idx / Kp, k = idx - n * Kp;
  const float v = (k < K) ? W[(size_t)k * Nfull + n] : 0.f;
  Wt[idx] = (_Float16)v;
}

__global__ void zero_u32(unsigned* __restrict__ p, int n) {
  const int i = blockIdx.x * 256 + threadIdx.x;
  if (i < n) p[i] = 0u;
}

// ---------------------------------------------------------------------------
// out[i] = [ pooled[batch_skip[i]] (1024) | x_skip[i] (256) ]   (f32, 1280/row)
// ---------------------------------------------------------------------------
__global__ __launch_bounds__(256) void assemble_out(
    const float* __restrict__ pooled, const float* __restrict__ x_skip,
    const int* __restrict__ batch_skip, float* __restrict__ out) {
  const int row = blockIdx.x;
  const int g = batch_skip[row];
  const float4* ps = (const float4*)(pooled + (size_t)g * N3);
  const float4* xs = (const float4*)(x_skip + (size_t)row * 256);
  float4* o = (float4*)(out + (size_t)row * 1280);
  const int t = threadIdx.x;
  o[t] = ps[t];                    // 256 float4 = 1024 floats
  if (t < 64) o[256 + t] = xs[t];  // 64 float4 = 256 floats
}

extern "C" void kernel_launch(void* const* d_in, const int* in_sizes, int n_in,
                              void* d_out, int out_size, void* d_ws, size_t ws_size,
                              hipStream_t stream) {
  // inputs: 0:x 1:pos 2:batch 3:x_skip 4:pos_skip 5:batch_skip
  //         6:W1 7:b1 8:W2 9:b2 10:W3 11:b3
  const float* x        = (const float*)d_in[0];
  const float* pos      = (const float*)d_in[1];
  const float* x_skip   = (const float*)d_in[3];
  const int*   bskip    = (const int*)d_in[5];
  const float* W1       = (const float*)d_in[6];
  const float* b1       = (const float*)d_in[7];
  const float* W2       = (const float*)d_in[8];
  const float* b2       = (const float*)d_in[9];
  const float* W3       = (const float*)d_in[10];
  const float* b3       = (const float*)d_in[11];
  float* out = (float*)d_out;

  // workspace layout
  char* ws = (char*)d_ws;
  unsigned*  pooled = (unsigned*)ws;                         // 16*1024 u32 (float bits)
  _Float16* W1t = (_Float16*)(ws + 65536);                   // [256][288]
  _Float16* W2t = W1t + 256 * K1P;                           // [512][256]
  _Float16* W3t = W2t + 512 * N1;                            // [1024][512]

  zero_u32<<<(16 * N3 + 255) / 256, 256, 0, stream>>>(pooled, 16 * N3);
  transpose_pad_f16<<<(256 * K1P + 255) / 256, 256, 0, stream>>>(W1, W1t, 259, 256, K1P);
  transpose_pad_f16<<<(512 * N1 + 255) / 256, 256, 0, stream>>>(W2, W2t, 256, 512, N1);
  transpose_pad_f16<<<(1024 * N2 + 255) / 256, 256, 0, stream>>>(W3, W3t, 512, 1024, N2);

  const size_t ldsBytes = (size_t)(TILE_M * K1P + TILE_M * N1 + TILE_M * N2) * sizeof(_Float16)
                        + (size_t)N3 * sizeof(unsigned);     // 274,432 B
  (void)hipFuncSetAttribute((const void*)fused_mlp_pool,
                            hipFuncAttributeMaxDynamicSharedMemorySize, (int)ldsBytes);
  fused_mlp_pool<<<NROWS / TILE_M, 256, ldsBytes, stream>>>(
      x, pos, W1t, b1, W2t, b2, W3t, b3, pooled);

  assemble_out<<<NROWS, 256, 0, stream>>>((const float*)pooled, x_skip, bskip, out);
}
